// GCNEncoder_86912958202425
// MI455X (gfx1250) — matrix operand order repaired and checked by reading, not compiled, not run on hardware
//
#include <hip/hip_runtime.h>
#include <hip/hip_fp16.h>

#define NN 200000
#define NE 400000
#define INDIM 128
#define HID 256
#define NL 3
#define NG 4096
#define BN_EPS 1e-5f

typedef __attribute__((ext_vector_type(16))) _Float16 v16h;
typedef __attribute__((ext_vector_type(8)))  float    v8f;

union AFrag { v16h v; _Float16 h[16]; };
union BFrag { v16h v; uint4 q[2]; };

__device__ __forceinline__ void atomAddF(float* p, float v) { unsafeAtomicAdd(p, v); }

// ---------------------------------------------------------------------------
// Generic f32 zero
__global__ void k_zero(float* __restrict__ p, long n) {
    long i = (long)blockIdx.x * blockDim.x + threadIdx.x;
    if (i < n) p[i] = 0.f;
}

// Convert + transpose weights: src is [K][N] f32 row-major, dst is [N][K] f16
__global__ void k_cvt_transpose(const float* __restrict__ src, _Float16* __restrict__ dst,
                                int K, int N) {
    int idx = blockIdx.x * blockDim.x + threadIdx.x;
    if (idx >= K * N) return;
    int n = idx / K, k = idx % K;
    dst[idx] = (_Float16)src[k * N + n];
}

// Degree with self-loop baseline of 1
__global__ void k_deg_init(float* __restrict__ deg) {
    int i = blockIdx.x * blockDim.x + threadIdx.x;
    if (i < NN) deg[i] = 1.0f;
}
__global__ void k_deg_acc(const int* __restrict__ ei, float* __restrict__ deg) {
    int e = blockIdx.x * blockDim.x + threadIdx.x;
    if (e < NE) atomAddF(&deg[ei[NE + e]], 1.0f);   // dst row
}
__global__ void k_dinv(const float* __restrict__ deg, float* __restrict__ dinv) {
    int i = blockIdx.x * blockDim.x + threadIdx.x;
    if (i < NN) dinv[i] = rsqrtf(deg[i]);
}
__global__ void k_norm(const int* __restrict__ ei, const float* __restrict__ dinv,
                       float* __restrict__ normE) {
    int e = blockIdx.x * blockDim.x + threadIdx.x;
    if (e < NE) normE[e] = dinv[ei[e]] * dinv[ei[NE + e]];
}

// ---------------------------------------------------------------------------
// WMMA GEMM: C[M][HID] = A[M][K] (f32, cvt->f16) @ B[K][HID] with Bt=[HID][K] f16.
// One wave computes a 16x64 tile: grid = (M/16, HID/64), block = 32 (wave32).
__global__ __launch_bounds__(32)
void k_gemm_wmma(const float* __restrict__ A, const _Float16* __restrict__ Bt,
                 const float* __restrict__ bias, float* __restrict__ C,
                 int M, int K, int relu) {
    const int lane  = threadIdx.x;                 // 0..31
    const int m0    = blockIdx.x * 16;
    const int n0    = blockIdx.y * 64;
    const int row   = m0 + (lane & 15);
    const int kbA   = (lane >> 4) * 8;             // 16-bit A layout: K 0-7/16-23 | 8-15/24-31
    const int col16 = lane & 15;
    const int kbB   = (lane >> 4) * 8;             // B layout: K 0-7/16-23 | 8-15/24-31

    v8f zero8 = {0.f,0.f,0.f,0.f,0.f,0.f,0.f,0.f};
    v8f acc[4] = {zero8, zero8, zero8, zero8};

    const float* arow = A + (long)row * K;
    for (int k0 = 0; k0 < K; k0 += 32) {
        const float* ap = arow + k0 + kbA;
        __builtin_prefetch((const void*)(ap + 32), 0, 0);
        float4 f0 = *(const float4*)(ap);
        float4 f1 = *(const float4*)(ap + 4);
        float4 f2 = *(const float4*)(ap + 16);
        float4 f3 = *(const float4*)(ap + 20);
        AFrag af;
        af.h[0]=(_Float16)f0.x;  af.h[1]=(_Float16)f0.y;  af.h[2]=(_Float16)f0.z;  af.h[3]=(_Float16)f0.w;
        af.h[4]=(_Float16)f1.x;  af.h[5]=(_Float16)f1.y;  af.h[6]=(_Float16)f1.z;  af.h[7]=(_Float16)f1.w;
        af.h[8]=(_Float16)f2.x;  af.h[9]=(_Float16)f2.y;  af.h[10]=(_Float16)f2.z; af.h[11]=(_Float16)f2.w;
        af.h[12]=(_Float16)f3.x; af.h[13]=(_Float16)f3.y; af.h[14]=(_Float16)f3.z; af.h[15]=(_Float16)f3.w;

        #pragma unroll
        for (int t = 0; t < 4; ++t) {
            const _Float16* bp = Bt + (long)(n0 + t*16 + col16) * K + k0 + kbB;
            BFrag bf;
            bf.q[0] = *(const uint4*)(bp);
            bf.q[1] = *(const uint4*)(bp + 16);
            acc[t] = __builtin_amdgcn_wmma_f32_16x16x32_f16(
                         false, af.v, false, bf.v, (short)0, acc[t], false, false);
        }
    }

    // C/D layout: VGPR r, lanes 0-15 -> M=r, lanes 16-31 -> M=8+r; N = lane&15
    const int rbase = m0 + ((lane >> 4) * 8);
    #pragma unroll
    for (int t = 0; t < 4; ++t) {
        const int cc = n0 + t*16 + col16;
        const float b = bias ? bias[cc] : 0.f;
        #pragma unroll
        for (int r = 0; r < 8; ++r) {
            float v = acc[t][r] + b;
            if (relu) v = fmaxf(v, 0.f);
            C[(long)(rbase + r) * HID + cc] = v;
        }
    }
}

// ---------------------------------------------------------------------------
// Edge scatter: agg[dst] += m[src] * norm, incl. self-loops (e >= NE)
__global__ void k_scatter(const int* __restrict__ ei, const float* __restrict__ normE,
                          const float* __restrict__ dinv, const float* __restrict__ m,
                          float* __restrict__ agg) {
    long gid = (long)blockIdx.x * blockDim.x + threadIdx.x;
    const long total = (long)(NE + NN) * 64;
    if (gid >= total) return;
    int e = (int)(gid >> 6);
    int c = (int)(gid & 63) * 4;
    int s, d; float nrm;
    if (e < NE) { s = ei[e]; d = ei[NE + e]; nrm = normE[e]; }
    else        { s = d = e - NE; float x = dinv[s]; nrm = x * x; }
    float4 v = *(const float4*)(m + (long)s * HID + c);
    float* o = agg + (long)d * HID + c;
    atomAddF(o + 0, v.x * nrm);
    atomAddF(o + 1, v.y * nrm);
    atomAddF(o + 2, v.z * nrm);
    atomAddF(o + 3, v.w * nrm);
}

// BN(eval) + ReLU + residual:  h = relu((agg + b - mean)*rsqrt(var+eps)*gamma + beta) + h
__global__ void k_bn_relu_res(const float* __restrict__ agg, float* __restrict__ h,
                              const float* __restrict__ cb, const float* __restrict__ gamma,
                              const float* __restrict__ beta, const float* __restrict__ mean,
                              const float* __restrict__ var, int l) {
    long idx = (long)blockIdx.x * blockDim.x + threadIdx.x;
    if (idx >= (long)NN * HID) return;
    int c = (int)(idx & (HID - 1));
    int o = l * HID + c;
    float a  = agg[idx] + cb[o];
    float bn = (a - mean[o]) * rsqrtf(var[o] + BN_EPS) * gamma[o] + beta[o];
    h[idx] = fmaxf(bn, 0.f) + h[idx];
}

// Global mean pool
__global__ void k_counts(const int* __restrict__ batch, float* __restrict__ counts) {
    int i = blockIdx.x * blockDim.x + threadIdx.x;
    if (i < NN) atomAddF(&counts[batch[i]], 1.0f);
}
__global__ void k_pool(const int* __restrict__ batch, const float* __restrict__ h,
                       float* __restrict__ pooled) {
    long gid = (long)blockIdx.x * blockDim.x + threadIdx.x;
    if (gid >= (long)NN * 64) return;
    int i = (int)(gid >> 6);
    int c = (int)(gid & 63) * 4;
    int g = batch[i];
    float4 v = *(const float4*)(h + (long)i * HID + c);
    float* o = pooled + (long)g * HID + c;
    atomAddF(o + 0, v.x); atomAddF(o + 1, v.y);
    atomAddF(o + 2, v.z); atomAddF(o + 3, v.w);
}
__global__ void k_pool_div(float* __restrict__ pooled, const float* __restrict__ counts) {
    int idx = blockIdx.x * blockDim.x + threadIdx.x;
    if (idx >= NG * HID) return;
    pooled[idx] = pooled[idx] / fmaxf(counts[idx >> 8], 1.0f);
}

// ---------------------------------------------------------------------------
static inline size_t alignup(size_t x) { return (x + 255) & ~(size_t)255; }

extern "C" void kernel_launch(void* const* d_in, const int* in_sizes, int n_in,
                              void* d_out, int out_size, void* d_ws, size_t ws_size,
                              hipStream_t stream) {
    const float* x       = (const float*)d_in[0];
    const int*   ei      = (const int*)  d_in[1];
    const int*   batch   = (const int*)  d_in[2];
    const float* emb_W   = (const float*)d_in[3];
    const float* emb_b   = (const float*)d_in[4];
    const float* conv_W  = (const float*)d_in[5];
    const float* conv_b  = (const float*)d_in[6];
    const float* bn_g    = (const float*)d_in[7];
    const float* bn_b    = (const float*)d_in[8];
    const float* bn_m    = (const float*)d_in[9];
    const float* bn_v    = (const float*)d_in[10];
    const float* out_W1  = (const float*)d_in[11];
    const float* out_b1  = (const float*)d_in[12];
    const float* out_W2  = (const float*)d_in[13];
    const float* out_b2  = (const float*)d_in[14];
    float* out = (float*)d_out;

    // workspace carve-up
    char* ws = (char*)d_ws;
    size_t off = 0;
    auto take = [&](size_t bytes) { char* p = ws + off; off += alignup(bytes); return p; };
    float*    h      = (float*)   take((size_t)NN * HID * 4);
    float*    mbuf   = (float*)   take((size_t)NN * HID * 4);
    float*    agg    = (float*)   take((size_t)NN * HID * 4);
    float*    deg    = (float*)   take((size_t)NN * 4);
    float*    dinv   = (float*)   take((size_t)NN * 4);
    float*    normE  = (float*)   take((size_t)NE * 4);
    float*    pooled = (float*)   take((size_t)NG * HID * 4);
    float*    counts = (float*)   take((size_t)NG * 4);
    float*    t1     = (float*)   take((size_t)NG * HID * 4);
    _Float16* embT   = (_Float16*)take((size_t)INDIM * HID * 2);
    _Float16* convT  = (_Float16*)take((size_t)NL * HID * HID * 2);
    _Float16* w1T    = (_Float16*)take((size_t)HID * HID * 2);
    _Float16* w2T    = (_Float16*)take((size_t)HID * HID * 2);

    const int B = 256;
    // 1) weight prep (f32 -> f16, transpose to [N][K])
    k_cvt_transpose<<<(INDIM*HID + B-1)/B, B, 0, stream>>>(emb_W, embT, INDIM, HID);
    for (int l = 0; l < NL; ++l)
        k_cvt_transpose<<<(HID*HID + B-1)/B, B, 0, stream>>>(conv_W + (size_t)l*HID*HID,
                                                             convT + (size_t)l*HID*HID, HID, HID);
    k_cvt_transpose<<<(HID*HID + B-1)/B, B, 0, stream>>>(out_W1, w1T, HID, HID);
    k_cvt_transpose<<<(HID*HID + B-1)/B, B, 0, stream>>>(out_W2, w2T, HID, HID);

    // 2) GCN normalization
    k_deg_init<<<(NN + B-1)/B, B, 0, stream>>>(deg);
    k_deg_acc <<<(NE + B-1)/B, B, 0, stream>>>(ei, deg);
    k_dinv    <<<(NN + B-1)/B, B, 0, stream>>>(deg, dinv);
    k_norm    <<<(NE + B-1)/B, B, 0, stream>>>(ei, dinv, normE);

    // 3) embedding GEMM: h = x @ emb_W + emb_b
    k_gemm_wmma<<<dim3(NN/16, HID/64), 32, 0, stream>>>(x, embT, emb_b, h, NN, INDIM, 0);

    // 4) GCN layers
    const long nh = (long)NN * HID;
    const long scat_threads = (long)(NE + NN) * 64;
    for (int l = 0; l < NL; ++l) {
        k_gemm_wmma<<<dim3(NN/16, HID/64), 32, 0, stream>>>(h, convT + (size_t)l*HID*HID,
                                                            nullptr, mbuf, NN, HID, 0);
        k_zero<<<(unsigned)((nh + B-1)/B), B, 0, stream>>>(agg, nh);
        k_scatter<<<(unsigned)((scat_threads + B-1)/B), B, 0, stream>>>(ei, normE, dinv, mbuf, agg);
        k_bn_relu_res<<<(unsigned)((nh + B-1)/B), B, 0, stream>>>(agg, h, conv_b,
                                                                  bn_g, bn_b, bn_m, bn_v, l);
    }

    // 5) global mean pool
    k_zero<<<((long)NG*HID + B-1)/B, B, 0, stream>>>(pooled, (long)NG*HID);
    k_zero<<<(NG + B-1)/B, B, 0, stream>>>(counts, NG);
    k_counts<<<(NN + B-1)/B, B, 0, stream>>>(batch, counts);
    k_pool<<<(unsigned)(((long)NN*64 + B-1)/B), B, 0, stream>>>(batch, h, pooled);
    k_pool_div<<<(NG*HID + B-1)/B, B, 0, stream>>>(pooled, counts);

    // 6) output MLP
    k_gemm_wmma<<<dim3(NG/16, HID/64), 32, 0, stream>>>(pooled, w1T, out_b1, t1, NG, HID, 1);
    k_gemm_wmma<<<dim3(NG/16, HID/64), 32, 0, stream>>>(t1, w2T, out_b2, out, NG, HID, 0);

    (void)in_sizes; (void)n_in; (void)out_size; (void)ws_size;
}
